// QuantumDQN_70377334112276
// MI455X (gfx1250) — compile-verified, matrix operand order checked by reading, not run on hardware
//
#include <hip/hip_runtime.h>
#include <hip/hip_bf16.h>
#include <stdint.h>

// ---------------- problem constants ----------------
#define MDIM 8192
#define KDIM 7056
#define KP   7072        // K padded to a multiple of TK (221*32); pad columns are zero
#define NDIM 512
#define FEAT8 8
#define HID 128
#define NACT 6
#define PI_F 3.14159265358979323846f
#define RH 0.70710678118654752f

typedef __attribute__((ext_vector_type(8)))  float  v8f;
typedef __attribute__((ext_vector_type(16))) __bf16 v16bf;

union FragU { uint4 u[2]; v16bf v; };

__device__ __forceinline__ unsigned short f2bf(float f) {
    unsigned int u = __float_as_uint(f);
    unsigned int r = u + 0x7FFFu + ((u >> 16) & 1u);   // round-to-nearest-even
    return (unsigned short)(r >> 16);
}

// ====== K0a: convert x (M x KDIM f32) -> xbf (M x KP bf16, zero-padded tail) ======
__global__ __launch_bounds__(256) void convert_pad_x(const float* __restrict__ x,
                                                     unsigned short* __restrict__ xbf) {
    const int col = blockIdx.x * 1024 + threadIdx.x * 4;
    const size_t row = blockIdx.y;
    if (col < KP) {
        float4 v = make_float4(0.f, 0.f, 0.f, 0.f);
        if (col < KDIM) v = *(const float4*)(x + row * KDIM + col);   // KDIM%4==0: no straddle
        unsigned int lo = (unsigned)f2bf(v.x) | ((unsigned)f2bf(v.y) << 16);
        unsigned int hi = (unsigned)f2bf(v.z) | ((unsigned)f2bf(v.w) << 16);
        *(uint2*)(xbf + row * KP + col) = make_uint2(lo, hi);
    }
}

// ====== K0b: transpose + convert pre_W1 (KDIM x N f32) -> W1t (N x KP bf16, zero-padded) ======
__global__ __launch_bounds__(256) void transposeW1(const float* __restrict__ w1,
                                                   unsigned short* __restrict__ w1t) {
    __shared__ float tile[16][17];
    const int tx = threadIdx.x & 15, ty = threadIdx.x >> 4;
    const int k0 = blockIdx.x * 16, n0 = blockIdx.y * 16;
    tile[ty][tx] = (k0 + ty < KDIM) ? w1[(size_t)(k0 + ty) * NDIM + n0 + tx] : 0.f;
    __syncthreads();
    w1t[(size_t)(n0 + ty) * KP + k0 + tx] = f2bf(tile[tx][ty]);
}

// ================= K1: h = relu(x @ W1 + b1)  [8192x7056x512, bf16 WMMA, f32 accum] ==============
#define TM 128
#define TN 128
#define TK 32
#define NKB (KP / TK)   // 221, exact — no tail

// bf16 row-major tile loader: 128 rows x 32 k, one uint4 (8 bf16) per thread-chunk, no predicates
__device__ __forceinline__ void g2r_T(const unsigned short* __restrict__ src, int r0, int k0,
                                      int t, uint4 (&reg)[2]) {
    #pragma unroll
    for (int i = 0; i < 2; ++i) {
        int idx = t + i * 256;
        int r = idx >> 2, c = idx & 3;
        reg[i] = *(const uint4*)(src + (size_t)(r0 + r) * KP + k0 + c * 8);
    }
}
__device__ __forceinline__ void r2l_T(unsigned short* dst, int t, const uint4 (&reg)[2]) {
    #pragma unroll
    for (int i = 0; i < 2; ++i) {
        int idx = t + i * 256;
        int r = idx >> 2, c = idx & 3;
        *(uint4*)&dst[r * TK + c * 8] = reg[i];
    }
}

__global__ __launch_bounds__(256) void gemm1(const unsigned short* __restrict__ xbf,
                                             const unsigned short* __restrict__ w1t,
                                             const float* __restrict__ b1,
                                             float* __restrict__ h) {
    __shared__ unsigned short As[2][TM * TK];
    __shared__ unsigned short Bs[2][TN * TK];
    const int t = threadIdx.x;
    const int lane = t & 31, wave = t >> 5;
    const int wm = wave & 3, wn = wave >> 2;           // wave tile: rows wm*32, cols wn*64
    const int m0 = blockIdx.y * TM, n0 = blockIdx.x * TN;
    const int hl = lane & 15, hi = lane >> 4;

    v8f acc[2][4];
    #pragma unroll
    for (int i = 0; i < 2; ++i)
        #pragma unroll
        for (int j = 0; j < 4; ++j)
            acc[i][j] = (v8f){0.f, 0.f, 0.f, 0.f, 0.f, 0.f, 0.f, 0.f};

    uint4 areg[2], breg[2];

    g2r_T(xbf, m0, 0, t, areg);
    g2r_T(w1t, n0, 0, t, breg);
    r2l_T(As[0], t, areg);
    r2l_T(Bs[0], t, breg);
    __syncthreads();

    for (int kb = 0; kb < NKB; ++kb) {
        const int cur = kb & 1, nxt = cur ^ 1;
        if (kb + 1 < NKB) {
            g2r_T(xbf, m0, (kb + 1) * TK, t, areg);
            g2r_T(w1t, n0, (kb + 1) * TK, t, breg);
        }
        FragU afrag[2], bfrag[4];
        #pragma unroll
        for (int i = 0; i < 2; ++i) {
            int row = wm * 32 + i * 16 + hl;
            const unsigned short* p = &As[cur][row * TK + hi * 8];
            afrag[i].u[0] = *(const uint4*)(p);
            afrag[i].u[1] = *(const uint4*)(p + 16);
        }
        #pragma unroll
        for (int j = 0; j < 4; ++j) {
            int n = wn * 64 + j * 16 + hl;
            const unsigned short* p = &Bs[cur][n * TK + hi * 8];
            bfrag[j].u[0] = *(const uint4*)(p);
            bfrag[j].u[1] = *(const uint4*)(p + 16);
        }
        #pragma unroll
        for (int i = 0; i < 2; ++i)
            #pragma unroll
            for (int j = 0; j < 4; ++j)
                acc[i][j] = __builtin_amdgcn_wmma_f32_16x16x32_bf16(
                    false, afrag[i].v, false, bfrag[j].v, (short)0, acc[i][j], false, false);
        if (kb + 1 < NKB) {
            r2l_T(As[nxt], t, areg);
            r2l_T(Bs[nxt], t, breg);
        }
        __syncthreads();
    }

    // epilogue: bias + relu, D layout: vgpr r -> M = r + 8*hi, N = lane&15
    #pragma unroll
    for (int j = 0; j < 4; ++j) {
        const int col = n0 + wn * 64 + j * 16 + hl;
        const float bias = b1[col];
        #pragma unroll
        for (int i = 0; i < 2; ++i) {
            const int rbase = m0 + wm * 32 + i * 16 + 8 * hi;
            #pragma unroll
            for (int r = 0; r < 8; ++r) {
                float v = acc[i][j][r] + bias;
                h[(size_t)(rbase + r) * NDIM + col] = v > 0.f ? v : 0.f;
            }
        }
    }
}

// ================= K2: feats8 = tanh(h @ pre_W2[:, :8] + b2[:8]) =================
__global__ __launch_bounds__(256) void feat8_kernel(const float* __restrict__ h,
                                                    const float* __restrict__ w2,
                                                    const float* __restrict__ b2,
                                                    float* __restrict__ f8) {
    __shared__ float w2s[NDIM * FEAT8];
    const int t = threadIdx.x;
    #pragma unroll
    for (int i = 0; i < 16; ++i) {
        int idx = t + i * 256;
        w2s[idx] = w2[(size_t)(idx >> 3) * 256 + (idx & 7)];
    }
    __syncthreads();
    const int row = blockIdx.x * 32 + (t >> 3);
    const int col = t & 7;
    const float* hr = h + (size_t)row * NDIM;
    float a = b2[col];
    #pragma unroll 4
    for (int k = 0; k < NDIM; ++k) a += hr[k] * w2s[k * FEAT8 + col];
    f8[row * FEAT8 + col] = tanhf(a);
}

// ================= K3: 8-qubit statevector sim + post-MLP (one sample per wave32) =================
// amplitude index i (0..255): i = lane*8 + r. bits 0..2 live in registers (r), bits 3..7 in lanes.
// qubit q acts on bit b = 7-q.

template <int B>
__device__ __forceinline__ void gate1(float (&sr)[8], float (&si)[8],
                                      float a00r, float a00i, float a01r, float a01i,
                                      float a10r, float a10i, float a11r, float a11i, int lane) {
    if constexpr (B < 3) {
        constexpr int m = 1 << B;
        #pragma unroll
        for (int r = 0; r < 8; ++r) {
            if ((r & m) == 0) {
                const int r1 = r | m;
                float xr = sr[r], xi = si[r], yr = sr[r1], yi = si[r1];
                sr[r]  = a00r * xr - a00i * xi + a01r * yr - a01i * yi;
                si[r]  = a00r * xi + a00i * xr + a01r * yi + a01i * yr;
                sr[r1] = a10r * xr - a10i * xi + a11r * yr - a11i * yi;
                si[r1] = a10r * xi + a10i * xr + a11r * yi + a11i * yr;
            }
        }
    } else {
        const int lm = 1 << (B - 3);
        const int myBit = (lane >> (B - 3)) & 1;
        #pragma unroll
        for (int r = 0; r < 8; ++r) {
            float pr = __shfl_xor(sr[r], lm, 32);
            float pi = __shfl_xor(si[r], lm, 32);
            if (myBit == 0) {
                float xr = sr[r], xi = si[r];
                sr[r] = a00r * xr - a00i * xi + a01r * pr - a01i * pi;
                si[r] = a00r * xi + a00i * xr + a01r * pi + a01i * pr;
            } else {
                float yr = sr[r], yi = si[r];
                sr[r] = a10r * pr - a10i * pi + a11r * yr - a11i * yi;
                si[r] = a10r * pi + a10i * pr + a11r * yi + a11i * yr;
            }
        }
    }
}

template <int BC, int BT>
__device__ __forceinline__ void cnot_g(float (&sr)[8], float (&si)[8], int lane) {
    if constexpr (BC < 3 && BT < 3) {
        #pragma unroll
        for (int r = 0; r < 8; ++r) {
            if (((r >> BC) & 1) && !((r >> BT) & 1)) {
                const int r1 = r | (1 << BT);
                float tr = sr[r]; sr[r] = sr[r1]; sr[r1] = tr;
                float ti = si[r]; si[r] = si[r1]; si[r1] = ti;
            }
        }
    } else if constexpr (BC >= 3 && BT >= 3) {
        const int ctrl = (lane >> (BC - 3)) & 1;
        const int lm = 1 << (BT - 3);
        #pragma unroll
        for (int r = 0; r < 8; ++r) {
            float tr = __shfl_xor(sr[r], lm, 32);
            float ti = __shfl_xor(si[r], lm, 32);
            if (ctrl) { sr[r] = tr; si[r] = ti; }
        }
    } else {  // BC >= 3 (lane bit), BT < 3 (register bit)
        const int ctrl = (lane >> (BC - 3)) & 1;
        #pragma unroll
        for (int r = 0; r < 8; ++r) {
            if (((r >> BT) & 1) == 0) {
                const int r1 = r | (1 << BT);
                if (ctrl) {
                    float tr = sr[r]; sr[r] = sr[r1]; sr[r1] = tr;
                    float ti = si[r]; si[r] = si[r1]; si[r1] = ti;
                }
            }
        }
    }
}

template <int Q>
__device__ __forceinline__ void init_q(float (&sr)[8], float (&si)[8], const float* ang, int lane) {
    constexpr int B = 7 - Q;
    gate1<B>(sr, si, RH, 0.f, RH, 0.f, RH, 0.f, -RH, 0.f, lane);   // H
    float th = ang[Q] * PI_F * 0.5f;
    float c = cosf(th), s = sinf(th);
    gate1<B>(sr, si, c, 0.f, -s, 0.f, s, 0.f, c, 0.f, lane);       // RY(angle*pi)
}

template <int Q>
__device__ __forceinline__ void rot_q(float (&sr)[8], float (&si)[8], const float* qp, int lane) {
    constexpr int B = 7 - Q;
    float cx = cosf(qp[0] * 0.5f), sx = sinf(qp[0] * 0.5f);
    gate1<B>(sr, si, cx, 0.f, 0.f, -sx, 0.f, -sx, cx, 0.f, lane);  // RX
    float cy = cosf(qp[1] * 0.5f), sy = sinf(qp[1] * 0.5f);
    gate1<B>(sr, si, cy, 0.f, -sy, 0.f, sy, 0.f, cy, 0.f, lane);   // RY
    float cz = cosf(qp[2] * 0.5f), sz = sinf(qp[2] * 0.5f);
    gate1<B>(sr, si, cz, -sz, 0.f, 0.f, 0.f, 0.f, cz, sz, lane);   // RZ
}

__global__ __launch_bounds__(256) void quantum_mlp(const float* __restrict__ f8,
                                                   const float* __restrict__ qparams,
                                                   const float* __restrict__ postW1,
                                                   const float* __restrict__ postB1,
                                                   const float* __restrict__ postW2,
                                                   const float* __restrict__ postB2,
                                                   const float* __restrict__ postW3,
                                                   const float* __restrict__ postB3,
                                                   float* __restrict__ out) {
    __shared__ float sW1[FEAT8 * HID];          // 8x128
    __shared__ float sW2[HID * (HID / 2)];      // 128x64
    __shared__ float sW3[(HID / 2) * NACT + 2]; // 64x6
    __shared__ float sB1[HID], sB2[HID / 2], sB3[8];
    __shared__ float h2s[8 * HID], h3s[8 * (HID / 2)];

    const int t = threadIdx.x, lane = t & 31, wave = t >> 5;
    for (int i = t; i < FEAT8 * HID; i += 256)      sW1[i] = postW1[i];
    for (int i = t; i < HID * (HID / 2); i += 256)  sW2[i] = postW2[i];
    for (int i = t; i < (HID / 2) * NACT; i += 256) sW3[i] = postW3[i];
    if (t < HID)     sB1[t] = postB1[t];
    if (t < HID / 2) sB2[t] = postB2[t];
    if (t < NACT)    sB3[t] = postB3[t];
    __syncthreads();

    const int sample = blockIdx.x * 8 + wave;
    float ang[8];
    #pragma unroll
    for (int q = 0; q < 8; ++q) ang[q] = f8[sample * 8 + q];

    float sr[8], si[8];
    #pragma unroll
    for (int r = 0; r < 8; ++r) { sr[r] = 0.f; si[r] = 0.f; }
    if (lane == 0) sr[0] = 1.f;

    init_q<0>(sr, si, ang, lane); init_q<1>(sr, si, ang, lane);
    init_q<2>(sr, si, ang, lane); init_q<3>(sr, si, ang, lane);
    init_q<4>(sr, si, ang, lane); init_q<5>(sr, si, ang, lane);
    init_q<6>(sr, si, ang, lane); init_q<7>(sr, si, ang, lane);

    for (int l = 0; l < 3; ++l) {
        // even CNOT pairs (0,1)(2,3)(4,5)(6,7) ; bits: (7,6)(5,4)(3,2)(1,0)
        cnot_g<7, 6>(sr, si, lane);
        cnot_g<5, 4>(sr, si, lane);
        cnot_g<3, 2>(sr, si, lane);
        cnot_g<1, 0>(sr, si, lane);
        // odd CNOT pairs (1,2)(3,4)(5,6) ; bits: (6,5)(4,3)(2,1)
        cnot_g<6, 5>(sr, si, lane);
        cnot_g<4, 3>(sr, si, lane);
        cnot_g<2, 1>(sr, si, lane);
        const float* qp = qparams + l * 24;
        rot_q<0>(sr, si, qp + 0,  lane); rot_q<1>(sr, si, qp + 3,  lane);
        rot_q<2>(sr, si, qp + 6,  lane); rot_q<3>(sr, si, qp + 9,  lane);
        rot_q<4>(sr, si, qp + 12, lane); rot_q<5>(sr, si, qp + 15, lane);
        rot_q<6>(sr, si, qp + 18, lane); rot_q<7>(sr, si, qp + 21, lane);
    }

    // Z expectations per qubit q (bit 7-q)
    float ev[8];
    #pragma unroll
    for (int q = 0; q < 8; ++q) ev[q] = 0.f;
    #pragma unroll
    for (int r = 0; r < 8; ++r) {
        float p = sr[r] * sr[r] + si[r] * si[r];
        #pragma unroll
        for (int q = 0; q < 8; ++q) {
            const int b = 7 - q;
            int bit = (b < 3) ? ((r >> b) & 1) : ((lane >> (b - 3)) & 1);
            ev[q] += bit ? -p : p;
        }
    }
    #pragma unroll
    for (int q = 0; q < 8; ++q) {
        #pragma unroll
        for (int off = 16; off >= 1; off >>= 1) ev[q] += __shfl_xor(ev[q], off, 32);
    }

    // fused post-MLP: 8 -> 128 -> 64 -> 6
    float* mH2 = h2s + wave * HID;
    float* mH3 = h3s + wave * (HID / 2);
    #pragma unroll
    for (int jj = 0; jj < 4; ++jj) {
        int j = lane + jj * 32;
        float a = sB1[j];
        #pragma unroll
        for (int q = 0; q < 8; ++q) a += ev[q] * sW1[q * HID + j];
        mH2[j] = a > 0.f ? a : 0.f;
    }
    __syncthreads();
    #pragma unroll
    for (int kk = 0; kk < 2; ++kk) {
        int j2 = lane + kk * 32;
        float a = sB2[j2];
        for (int j = 0; j < HID; ++j) a += mH2[j] * sW2[j * (HID / 2) + j2];
        mH3[j2] = a > 0.f ? a : 0.f;
    }
    __syncthreads();
    if (lane < NACT) {
        float a = sB3[lane];
        #pragma unroll
        for (int j2 = 0; j2 < HID / 2; ++j2) a += mH3[j2] * sW3[j2 * NACT + lane];
        out[sample * NACT + lane] = a;
    }
}

// ================= launcher =================
extern "C" void kernel_launch(void* const* d_in, const int* in_sizes, int n_in,
                              void* d_out, int out_size, void* d_ws, size_t ws_size,
                              hipStream_t stream) {
    const float* x      = (const float*)d_in[0];
    const float* preW1  = (const float*)d_in[1];
    const float* preB1  = (const float*)d_in[2];
    const float* preW2  = (const float*)d_in[3];
    const float* preB2  = (const float*)d_in[4];
    const float* qparam = (const float*)d_in[5];
    const float* pW1    = (const float*)d_in[6];
    const float* pB1    = (const float*)d_in[7];
    const float* pW2    = (const float*)d_in[8];
    const float* pB2    = (const float*)d_in[9];
    const float* pW3    = (const float*)d_in[10];
    const float* pB3    = (const float*)d_in[11];
    float* out = (float*)d_out;

    char* ws = (char*)d_ws;
    size_t off = 0;
    float* hbuf = (float*)(ws + off);                  off += (size_t)MDIM * NDIM * 4; // 16.8 MB
    unsigned short* w1t = (unsigned short*)(ws + off); off += (size_t)NDIM * KP * 2;   // 7.2 MB
    float* f8 = (float*)(ws + off);                    off += (size_t)MDIM * FEAT8 * 4;
    unsigned short* xbf = (unsigned short*)(ws + off);                                 // 115.9 MB

    convert_pad_x<<<dim3((KP + 1023) / 1024, MDIM), 256, 0, stream>>>(x, xbf);
    transposeW1<<<dim3(KP / 16, NDIM / 16), 256, 0, stream>>>(preW1, w1t);
    gemm1<<<dim3(NDIM / TN, MDIM / TM), 256, 0, stream>>>(xbf, w1t, preB1, hbuf);
    feat8_kernel<<<MDIM / 32, 256, 0, stream>>>(hbuf, preW2, preB2, f8);
    quantum_mlp<<<MDIM / 8, 256, 0, stream>>>(f8, qparam, pW1, pB1, pW2, pB2, pW3, pB3, out);
}
